// SpatialStructuralEmbedder_11347303596106
// MI455X (gfx1250) — matrix operand order repaired and checked by reading, not compiled, not run on hardware
//
#include <hip/hip_runtime.h>

#define N_PATCH 8192
#define N_TILES 128
#define N_TOT   8320
#define N_EDGES 1024
#define IN_DIM  384
#define GNN_DIM 256
#define HEADS   4
#define DH      64
#define NEGINF  -3.0e38f

typedef __attribute__((ext_vector_type(16))) __bf16 v16bf;
typedef __attribute__((ext_vector_type(8)))  __bf16 v8bf;
typedef __attribute__((ext_vector_type(8)))  float  v8f;

__device__ __forceinline__ float lrelu(float x) { return x > 0.f ? x : 0.2f * x; }

// ---------------------------------------------------------------- converters
__global__ void build_xb(const float* __restrict__ x_nodes,
                         const float* __restrict__ rtok,
                         __bf16* __restrict__ xb) {
  int idx = blockIdx.x * blockDim.x + threadIdx.x;
  if (idx >= N_TOT * IN_DIM) return;
  int n = idx / IN_DIM, k = idx - n * IN_DIM;
  float v = (n < N_PATCH) ? x_nodes[idx] : rtok[k];
  xb[idx] = (__bf16)v;
}

// W [K][D] fp32 -> WT [D][K] bf16
__global__ void wT_f2bf(const float* __restrict__ W, __bf16* __restrict__ WT,
                        int K, int D) {
  int idx = blockIdx.x * blockDim.x + threadIdx.x;
  if (idx >= K * D) return;
  int k = idx / D, d = idx - k * D;
  WT[(size_t)d * K + k] = (__bf16)W[idx];
}

// Ht[e][n] = H[n][e]  (bf16 0/1), contiguous per-edge rows
__global__ void build_ht(const float* __restrict__ H, __bf16* __restrict__ Ht) {
  int idx = blockIdx.x * blockDim.x + threadIdx.x;
  if (idx >= N_TOT * N_EDGES) return;
  int n = idx / N_EDGES, e = idx - n * N_EDGES;
  Ht[(size_t)e * N_TOT + n] = (__bf16)H[idx];
}

// tiled fp32 [R][C] -> bf16 [C][R] transpose via LDS
__global__ __launch_bounds__(256) void transpose_f2bf(
    const float* __restrict__ src, __bf16* __restrict__ dst, int R, int C) {
  __shared__ float tile[32][33];
  int r0 = blockIdx.y * 32, c0 = blockIdx.x * 32;
  int tx = threadIdx.x & 31, ty = threadIdx.x >> 5;
  for (int i = ty; i < 32; i += 8)
    tile[i][tx] = src[(size_t)(r0 + i) * C + (c0 + tx)];
  __syncthreads();
  for (int i = ty; i < 32; i += 8)
    dst[(size_t)(c0 + i) * R + (r0 + tx)] = (__bf16)tile[tx][i];
}

// ------------------------------------------------------------ projection GEMM
// one wave -> 16 rows x 64 cols; A row-major K-contig, B = WT K-contig.
__global__ __launch_bounds__(128) void proj_gemm(
    const __bf16* __restrict__ A, int K,          // [N_TOT, K]
    const __bf16* __restrict__ WT,                // [GNN_DIM][K]
    const float* __restrict__ nemb,               // [4, GNN_DIM]
    const int* __restrict__ ntype,
    float* __restrict__ h, __bf16* __restrict__ hb) {
  int wave = threadIdx.x >> 5, lane = threadIdx.x & 31;
  int gid = blockIdx.x * 4 + wave;                // 520*4 strips
  int tm = gid >> 2, cg = gid & 3;
  int row0 = tm * 16, col0 = cg * 64;
  int nl = lane & 15;
  int koff  = (lane & 16) ? 16 : 0;               // A fragment k offset
  int kboff = (lane & 16) ? 16 : 0;               // B fragment k offset
  const __bf16* arow = A + (size_t)(row0 + nl) * K;
  v8f acc[4] = {};
  for (int k0 = 0; k0 < K; k0 += 32) {
    v8bf alo = *(const v8bf*)(arow + k0 + koff);
    v8bf ahi = *(const v8bf*)(arow + k0 + 8 + koff);
    v16bf af = __builtin_shufflevector(alo, ahi,
                 0,1,2,3,4,5,6,7,8,9,10,11,12,13,14,15);
#pragma unroll
    for (int t = 0; t < 4; ++t) {
      v16bf bfv = *(const v16bf*)(WT + (size_t)(col0 + t * 16 + nl) * K + k0 + kboff);
      acc[t] = __builtin_amdgcn_wmma_f32_16x16x32_bf16(false, af, false, bfv,
                                                       (short)0, acc[t], false, false);
    }
  }
  int rbase = (lane & 16) ? 8 : 0;
#pragma unroll
  for (int r = 0; r < 8; ++r) {
    int rr = row0 + r + rbase;
    int ntb = ntype[rr] * GNN_DIM;
#pragma unroll
    for (int t = 0; t < 4; ++t) {
      int cc = col0 + t * 16 + nl;
      float v = acc[t][r] + nemb[ntb + cc];
      h[(size_t)rr * GNN_DIM + cc]  = v;
      hb[(size_t)rr * GNN_DIM + cc] = (__bf16)v;
    }
  }
}

// ------------------------------------------------- attention score kernels
// head-major outputs: s[h][N]
__global__ void score_kernel(const float* __restrict__ h,
                             const float* __restrict__ asrc,
                             const float* __restrict__ adst,
                             float* __restrict__ s_src, float* __restrict__ s_dst) {
  int idx = blockIdx.x * blockDim.x + threadIdx.x;
  if (idx >= HEADS * N_TOT) return;
  int hh = idx / N_TOT, n = idx - hh * N_TOT;
  const float* base = h + (size_t)n * GNN_DIM + hh * DH;
  float s1 = 0.f, s2 = 0.f;
  for (int d = 0; d < DH; ++d) {
    float v = base[d];
    s1 += v * asrc[hh * DH + d];
    s2 += v * adst[hh * DH + d];
  }
  s_src[idx] = s1; s_dst[idx] = s2;
}

__global__ void edge_score(const float* __restrict__ m,
                           const float* __restrict__ aedg,
                           float* __restrict__ s_edg) {
  int idx = blockIdx.x * blockDim.x + threadIdx.x;
  if (idx >= HEADS * N_EDGES) return;
  int hh = idx / N_EDGES, e = idx - hh * N_EDGES;
  const float* base = m + (size_t)e * GNN_DIM + hh * DH;
  float s = 0.f;
  for (int d = 0; d < DH; ++d) s += base[d] * aedg[hh * DH + d];
  s_edg[idx] = s;
}

// -------------------------------------- softmax stats over nodes (per edge)
__global__ __launch_bounds__(256) void edge_stats(
    const __bf16* __restrict__ Ht, const float* __restrict__ s_src, // [H][N]
    const int* __restrict__ etype, const float* __restrict__ ebias, // [3][H]
    float* __restrict__ Mx, float* __restrict__ Z) {                // [H][E]
  __shared__ float red[HEADS * 256];
  int e = blockIdx.x, tid = threadIdx.x;
  int t = etype[e];
  float eb[HEADS];
#pragma unroll
  for (int hh = 0; hh < HEADS; ++hh) eb[hh] = ebias[t * HEADS + hh];
  const __bf16* hrow = Ht + (size_t)e * N_TOT;
  float lmax[HEADS];
#pragma unroll
  for (int hh = 0; hh < HEADS; ++hh) lmax[hh] = NEGINF;
  for (int n = tid; n < N_TOT; n += 256)
    if ((float)hrow[n] > 0.5f)
#pragma unroll
      for (int hh = 0; hh < HEADS; ++hh)
        lmax[hh] = fmaxf(lmax[hh], lrelu(s_src[hh * N_TOT + n] + eb[hh]));
#pragma unroll
  for (int hh = 0; hh < HEADS; ++hh) red[hh * 256 + tid] = lmax[hh];
  __syncthreads();
  for (int s = 128; s > 0; s >>= 1) {
    if (tid < s)
#pragma unroll
      for (int hh = 0; hh < HEADS; ++hh)
        red[hh * 256 + tid] = fmaxf(red[hh * 256 + tid], red[hh * 256 + tid + s]);
    __syncthreads();
  }
  float mx[HEADS];
#pragma unroll
  for (int hh = 0; hh < HEADS; ++hh) mx[hh] = red[hh * 256];
  if (tid < HEADS) Mx[tid * N_EDGES + e] = red[tid * 256];
  __syncthreads();
  float lsum[HEADS] = {0.f, 0.f, 0.f, 0.f};
  for (int n = tid; n < N_TOT; n += 256)
    if ((float)hrow[n] > 0.5f)
#pragma unroll
      for (int hh = 0; hh < HEADS; ++hh)
        lsum[hh] += __expf(lrelu(s_src[hh * N_TOT + n] + eb[hh]) - mx[hh]);
#pragma unroll
  for (int hh = 0; hh < HEADS; ++hh) red[hh * 256 + tid] = lsum[hh];
  __syncthreads();
  for (int s = 128; s > 0; s >>= 1) {
    if (tid < s)
#pragma unroll
      for (int hh = 0; hh < HEADS; ++hh)
        red[hh * 256 + tid] += red[hh * 256 + tid + s];
    __syncthreads();
  }
  if (tid < HEADS) Z[tid * N_EDGES + e] = red[tid * 256];
}

// -------------------------------------- softmax stats over edges (per node)
__global__ __launch_bounds__(128) void node_stats(
    const float* __restrict__ H, const float* __restrict__ s_dst,   // [H][N]
    const float* __restrict__ s_edg,                                // [H][E]
    float* __restrict__ Mx, float* __restrict__ Z) {                // [H][N]
  __shared__ float red[HEADS * 128];
  int n = blockIdx.x, tid = threadIdx.x;
  const float* hrow = H + (size_t)n * N_EDGES;
  float sd[HEADS];
#pragma unroll
  for (int hh = 0; hh < HEADS; ++hh) sd[hh] = s_dst[hh * N_TOT + n];
  float lmax[HEADS];
#pragma unroll
  for (int hh = 0; hh < HEADS; ++hh) lmax[hh] = NEGINF;
  for (int e = tid; e < N_EDGES; e += 128)
    if (hrow[e] > 0.f)
#pragma unroll
      for (int hh = 0; hh < HEADS; ++hh)
        lmax[hh] = fmaxf(lmax[hh], lrelu(sd[hh] + s_edg[hh * N_EDGES + e]));
#pragma unroll
  for (int hh = 0; hh < HEADS; ++hh) red[hh * 128 + tid] = lmax[hh];
  __syncthreads();
  for (int s = 64; s > 0; s >>= 1) {
    if (tid < s)
#pragma unroll
      for (int hh = 0; hh < HEADS; ++hh)
        red[hh * 128 + tid] = fmaxf(red[hh * 128 + tid], red[hh * 128 + tid + s]);
    __syncthreads();
  }
  float mx[HEADS];
#pragma unroll
  for (int hh = 0; hh < HEADS; ++hh) mx[hh] = red[hh * 128];
  if (tid < HEADS) Mx[tid * N_TOT + n] = red[tid * 128];
  __syncthreads();
  float lsum[HEADS] = {0.f, 0.f, 0.f, 0.f};
  for (int e = tid; e < N_EDGES; e += 128)
    if (hrow[e] > 0.f)
#pragma unroll
      for (int hh = 0; hh < HEADS; ++hh)
        lsum[hh] += __expf(lrelu(sd[hh] + s_edg[hh * N_EDGES + e]) - mx[hh]);
#pragma unroll
  for (int hh = 0; hh < HEADS; ++hh) red[hh * 128 + tid] = lsum[hh];
  __syncthreads();
  for (int s = 64; s > 0; s >>= 1) {
    if (tid < s)
#pragma unroll
      for (int hh = 0; hh < HEADS; ++hh)
        red[hh * 128 + tid] += red[hh * 128 + tid + s];
    __syncthreads();
  }
  if (tid < HEADS) Z[tid * N_TOT + n] = red[tid * 128];
}

// ---------------- fused node->edge aggregation: m[e,h,:] = softmax_n(A) @ h
// one wave -> 16 edges x 64 cols (one head); A generated in registers.
__global__ __launch_bounds__(128) void edge_aggr_gemm(
    const __bf16* __restrict__ Ht, const float* __restrict__ s_src,  // [H][N]
    const int* __restrict__ etype, const float* __restrict__ ebias,
    const float* __restrict__ Mx, const float* __restrict__ Z,       // [H][E]
    const __bf16* __restrict__ hT,                                   // [GNN_DIM][N_TOT]
    float* __restrict__ m) {
  int wave = threadIdx.x >> 5, lane = threadIdx.x & 31;
  int et = blockIdx.x * 4 + wave;                 // 64 e-tiles
  int head = blockIdx.y;
  int e0 = et * 16;
  int nl = lane & 15;
  int koff  = (lane & 16) ? 16 : 0;
  int kboff = (lane & 16) ? 16 : 0;
  int ea = e0 + nl;
  float eb = ebias[etype[ea] * HEADS + head];
  float mx = Mx[head * N_EDGES + ea];
  const __bf16* hrow = Ht + (size_t)ea * N_TOT;
  const float* srow  = s_src + (size_t)head * N_TOT;
  v8f acc[4] = {};
  for (int n0 = 0; n0 < N_TOT; n0 += 32) {        // 260 iterations
    __builtin_prefetch((const void*)(hrow + n0 + 128), 0, 1);
    v8bf wlo = *(const v8bf*)(hrow + n0 + koff);
    v8bf whi = *(const v8bf*)(hrow + n0 + 8 + koff);
    v8f  slo = *(const v8f*)(srow + n0 + koff);
    v8f  shi = *(const v8f*)(srow + n0 + 8 + koff);
    v16bf af;
#pragma unroll
    for (int i = 0; i < 8; ++i) {
      af[i]     = (__bf16)(((float)wlo[i] > 0.5f) ? __expf(lrelu(slo[i] + eb) - mx) : 0.f);
      af[i + 8] = (__bf16)(((float)whi[i] > 0.5f) ? __expf(lrelu(shi[i] + eb) - mx) : 0.f);
    }
#pragma unroll
    for (int t = 0; t < 4; ++t) {
      v16bf bfv = *(const v16bf*)(hT + (size_t)(head * DH + t * 16 + nl) * N_TOT
                                  + n0 + kboff);
      acc[t] = __builtin_amdgcn_wmma_f32_16x16x32_bf16(false, af, false, bfv,
                                                       (short)0, acc[t], false, false);
    }
  }
  int rbase = (lane & 16) ? 8 : 0;
#pragma unroll
  for (int t = 0; t < 4; ++t) {
    int cc = head * DH + t * 16 + nl;
#pragma unroll
    for (int r = 0; r < 8; ++r) {
      int ee = e0 + r + rbase;
      float z = Z[head * N_EDGES + ee];
      m[(size_t)ee * GNN_DIM + cc] = acc[t][r] / (z > 0.f ? z : 1.f);
    }
  }
}

// ---------------- fused edge->node aggregation + ELU + residual
__global__ __launch_bounds__(128) void node_aggr_gemm(
    const float* __restrict__ H, const float* __restrict__ s_dst,    // [H][N]
    const float* __restrict__ s_edg,                                 // [H][E]
    const float* __restrict__ Mx, const float* __restrict__ Z,       // [H][N]
    const __bf16* __restrict__ mT,                                   // [GNN_DIM][N_EDGES]
    const float* __restrict__ h_in,
    float* __restrict__ h_out, __bf16* __restrict__ hb_out) {
  int wave = threadIdx.x >> 5, lane = threadIdx.x & 31;
  int ntile = blockIdx.x * 4 + wave;              // 520 n-tiles
  int head = blockIdx.y;
  int n0 = ntile * 16;
  int nl = lane & 15;
  int koff  = (lane & 16) ? 16 : 0;
  int kboff = (lane & 16) ? 16 : 0;
  int na = n0 + nl;
  float sd = s_dst[head * N_TOT + na];
  float mx = Mx[head * N_TOT + na];
  const float* hrow  = H + (size_t)na * N_EDGES;
  const float* serow = s_edg + (size_t)head * N_EDGES;
  v8f acc[4] = {};
  for (int e0 = 0; e0 < N_EDGES; e0 += 32) {      // 32 iterations
    v8f wlo = *(const v8f*)(hrow + e0 + koff);
    v8f whi = *(const v8f*)(hrow + e0 + 8 + koff);
    v8f slo = *(const v8f*)(serow + e0 + koff);
    v8f shi = *(const v8f*)(serow + e0 + 8 + koff);
    v16bf af;
#pragma unroll
    for (int i = 0; i < 8; ++i) {
      af[i]     = (__bf16)((wlo[i] > 0.f) ? __expf(lrelu(sd + slo[i]) - mx) : 0.f);
      af[i + 8] = (__bf16)((whi[i] > 0.f) ? __expf(lrelu(sd + shi[i]) - mx) : 0.f);
    }
#pragma unroll
    for (int t = 0; t < 4; ++t) {
      v16bf bfv = *(const v16bf*)(mT + (size_t)(head * DH + t * 16 + nl) * N_EDGES
                                  + e0 + kboff);
      acc[t] = __builtin_amdgcn_wmma_f32_16x16x32_bf16(false, af, false, bfv,
                                                       (short)0, acc[t], false, false);
    }
  }
  int rbase = (lane & 16) ? 8 : 0;
#pragma unroll
  for (int t = 0; t < 4; ++t) {
    int cc = head * DH + t * 16 + nl;
#pragma unroll
    for (int r = 0; r < 8; ++r) {
      int nn = n0 + r + rbase;
      float z = Z[head * N_TOT + nn];
      float o = acc[t][r] / (z > 0.f ? z : 1.f);
      float el = o > 0.f ? o : (__expf(o) - 1.f);          // ELU
      float v = el + h_in[(size_t)nn * GNN_DIM + cc];      // residual
      h_out[(size_t)nn * GNN_DIM + cc]  = v;
      hb_out[(size_t)nn * GNN_DIM + cc] = (__bf16)v;
    }
  }
}

// ------------------------------------------------------------- final LN
__global__ __launch_bounds__(256) void ln_kernel(
    const float* __restrict__ h,
    const float* __restrict__ ng, const float* __restrict__ nb,
    const float* __restrict__ bg, const float* __restrict__ bb,
    float* __restrict__ out) {
  __shared__ float red[256];
  int n = blockIdx.x, tid = threadIdx.x;
  float x = h[(size_t)n * GNN_DIM + tid];
  red[tid] = x; __syncthreads();
  for (int s = 128; s > 0; s >>= 1) { if (tid < s) red[tid] += red[tid + s]; __syncthreads(); }
  float mean = red[0] / GNN_DIM; __syncthreads();
  float d = x - mean;
  red[tid] = d * d; __syncthreads();
  for (int s = 128; s > 0; s >>= 1) { if (tid < s) red[tid] += red[tid + s]; __syncthreads(); }
  float var = red[0] / GNN_DIM;
  const float* g = (n < N_PATCH) ? ng : bg;
  const float* b = (n < N_PATCH) ? nb : bb;
  out[(size_t)n * GNN_DIM + tid] = d * rsqrtf(var + 1e-5f) * g[tid] + b[tid];
}

// ---------------------------------------------------------------------------
extern "C" void kernel_launch(void* const* d_in, const int* in_sizes, int n_in,
                              void* d_out, int out_size, void* d_ws, size_t ws_size,
                              hipStream_t stream) {
  const float* x_nodes   = (const float*)d_in[0];
  const float* rtok      = (const float*)d_in[1];
  const int*   node_type = (const int*)d_in[2];
  const int*   edge_type = (const int*)d_in[3];
  const float* H         = (const float*)d_in[4];
  const float* W0        = (const float*)d_in[6];
  const float* W1        = (const float*)d_in[7];
  const float* node_emb  = (const float*)d_in[8];
  const float* a_src     = (const float*)d_in[9];
  const float* a_dst     = (const float*)d_in[10];
  const float* a_edge    = (const float*)d_in[11];
  const float* edge_bias = (const float*)d_in[12];
  const float* ngam      = (const float*)d_in[13];
  const float* nbet      = (const float*)d_in[14];
  const float* bgam      = (const float*)d_in[15];
  const float* bbet      = (const float*)d_in[16];
  float* out = (float*)d_out;

  size_t off = 0;
  auto carve = [&](size_t bytes) -> char* {
    char* p = (char*)d_ws + off;
    off += (bytes + 255) & ~(size_t)255;
    return p;
  };
  __bf16* xb   = (__bf16*)carve((size_t)N_TOT * IN_DIM * 2);
  __bf16* w0T  = (__bf16*)carve((size_t)IN_DIM * GNN_DIM * 2);   // [256][384]
  __bf16* w1T  = (__bf16*)carve((size_t)GNN_DIM * GNN_DIM * 2);  // [256][256]
  __bf16* Ht   = (__bf16*)carve((size_t)N_EDGES * N_TOT * 2);
  float*  hA   = (float*)carve((size_t)N_TOT * GNN_DIM * 4);
  __bf16* hbA  = (__bf16*)carve((size_t)N_TOT * GNN_DIM * 2);    // row-major
  __bf16* hAT  = (__bf16*)carve((size_t)GNN_DIM * N_TOT * 2);    // [256][N_TOT]
  float*  hB   = (float*)carve((size_t)N_TOT * GNN_DIM * 4);
  __bf16* hbB  = (__bf16*)carve((size_t)N_TOT * GNN_DIM * 2);
  float*  ssrc = (float*)carve((size_t)HEADS * N_TOT * 4);
  float*  sdst = (float*)carve((size_t)HEADS * N_TOT * 4);
  float*  sedg = (float*)carve((size_t)HEADS * N_EDGES * 4);
  float*  Mx1  = (float*)carve((size_t)HEADS * N_EDGES * 4);
  float*  Z1   = (float*)carve((size_t)HEADS * N_EDGES * 4);
  float*  Mx2  = (float*)carve((size_t)HEADS * N_TOT * 4);
  float*  Z2   = (float*)carve((size_t)HEADS * N_TOT * 4);
  float*  mmat = (float*)carve((size_t)N_EDGES * GNN_DIM * 4);
  __bf16* mT   = (__bf16*)carve((size_t)GNN_DIM * N_EDGES * 2);  // [256][1024]
  (void)ws_size; (void)in_sizes; (void)n_in; (void)out_size;

  build_xb<<<(N_TOT * IN_DIM + 255) / 256, 256, 0, stream>>>(x_nodes, rtok, xb);
  wT_f2bf<<<(IN_DIM * GNN_DIM + 255) / 256, 256, 0, stream>>>(W0, w0T, IN_DIM, GNN_DIM);
  wT_f2bf<<<(GNN_DIM * GNN_DIM + 255) / 256, 256, 0, stream>>>(W1, w1T, GNN_DIM, GNN_DIM);
  build_ht<<<(N_TOT * N_EDGES + 255) / 256, 256, 0, stream>>>(H, Ht);

  for (int layer = 0; layer < 2; ++layer) {
    const __bf16* inb = (layer == 0) ? xb : hbB;
    int K             = (layer == 0) ? IN_DIM : GNN_DIM;
    const __bf16* WT  = (layer == 0) ? w0T : w1T;
    const float* nemb = node_emb + layer * 4 * GNN_DIM;
    const float* asrc = a_src + layer * HEADS * DH;
    const float* adst = a_dst + layer * HEADS * DH;
    const float* aedg = a_edge + layer * HEADS * DH;
    const float* ebia = edge_bias + layer * 3 * HEADS;

    proj_gemm<<<(N_TOT / 16), 128, 0, stream>>>(inb, K, WT, nemb, node_type, hA, hbA);
    transpose_f2bf<<<dim3(GNN_DIM / 32, N_TOT / 32), 256, 0, stream>>>(
        hA, hAT, N_TOT, GNN_DIM);
    score_kernel<<<(HEADS * N_TOT + 255) / 256, 256, 0, stream>>>(hA, asrc, adst,
                                                                  ssrc, sdst);
    edge_stats<<<N_EDGES, 256, 0, stream>>>(Ht, ssrc, edge_type, ebia, Mx1, Z1);
    edge_aggr_gemm<<<dim3(N_EDGES / 64, HEADS), 128, 0, stream>>>(
        Ht, ssrc, edge_type, ebia, Mx1, Z1, hAT, mmat);
    transpose_f2bf<<<dim3(GNN_DIM / 32, N_EDGES / 32), 256, 0, stream>>>(
        mmat, mT, N_EDGES, GNN_DIM);
    edge_score<<<(HEADS * N_EDGES + 255) / 256, 256, 0, stream>>>(mmat, aedg, sedg);
    node_stats<<<N_TOT, 128, 0, stream>>>(H, sdst, sedg, Mx2, Z2);
    node_aggr_gemm<<<dim3(N_TOT / 64, HEADS), 128, 0, stream>>>(
        H, sdst, sedg, Mx2, Z2, mT, hA, hB, hbB);
  }

  ln_kernel<<<N_TOT, 256, 0, stream>>>(hB, ngam, nbet, bgam, bbet, out);
}
